// Fagcn_64501818851477
// MI455X (gfx1250) — matrix operand order repaired and checked by reading, not compiled, not run on hardware
//
#include <hip/hip_runtime.h>

typedef __attribute__((ext_vector_type(16))) _Float16 v16h;
typedef __attribute__((ext_vector_type(8)))  float    v8f;

#define HD   128
#define EPSW 0.3f

__device__ __forceinline__ void atomAddF(float* p, float v) {
    unsafeAtomicAdd(p, v);   // lowers to global_atomic_add_f32 (no CAS loop)
}

// ---------------- degree / norm ----------------
__global__ void deg_kernel(const int* __restrict__ dst, float* __restrict__ deg, int E) {
    int e = blockIdx.x * blockDim.x + threadIdx.x;
    if (e < E) atomAddF(&deg[dst[e]], 1.0f);
}

__global__ void norm_kernel(float* __restrict__ d, int N) {
    int i = blockIdx.x * blockDim.x + threadIdx.x;
    if (i < N) d[i] = rsqrtf(fmaxf(d[i], 1.0f));
}

// ---------------- WMMA projection: (16x128 tile of h) x (128x2 weights) ----------------
// One wave per 16-node tile. B columns: col0 = gw[0:H] (a_dst), col1 = gw[H:2H] (a_src).
__global__ void proj_kernel(const float* __restrict__ h,
                            const float* __restrict__ gw,   // 2H floats for this layer
                            float* __restrict__ adst, float* __restrict__ asrc,
                            int N) {
    int wave = (blockIdx.x * blockDim.x + threadIdx.x) >> 5;
    int lane = threadIdx.x & 31;
    int numTiles = (N + 15) >> 4;
    if (wave >= numTiles) return;              // wave-uniform exit; EXEC all-1 below

    int m   = lane & 15;
    int row = wave * 16 + m;
    if (row >= N) row = N - 1;                 // clamp (unused rows never stored)
    int khalf = (lane >> 4) << 4;              // A: 0 or 16
    int kbase = (lane >> 4) << 4;              // B: 0 or 16
    const float4* hrow = (const float4*)(h + (size_t)row * HD);
    int col = m;                               // B column owned by this lane pair

    v8f acc = {};
#pragma unroll
    for (int c = 0; c < 4; ++c) {
        int k0 = c * 32;
        // ---- A tile: element j = K = k0+khalf+j
        float4 f0 = hrow[(k0 + khalf) >> 2];
        float4 f1 = hrow[((k0 + khalf) >> 2) + 1];
        float4 f2 = hrow[((k0 + khalf) >> 2) + 2];
        float4 f3 = hrow[((k0 + khalf) >> 2) + 3];
        v16h a;
        a[0]=(_Float16)f0.x; a[1]=(_Float16)f0.y; a[2]=(_Float16)f0.z; a[3]=(_Float16)f0.w;
        a[4]=(_Float16)f1.x; a[5]=(_Float16)f1.y; a[6]=(_Float16)f1.z; a[7]=(_Float16)f1.w;
        a[8]=(_Float16)f2.x; a[9]=(_Float16)f2.y; a[10]=(_Float16)f2.z; a[11]=(_Float16)f2.w;
        a[12]=(_Float16)f3.x; a[13]=(_Float16)f3.y; a[14]=(_Float16)f3.z; a[15]=(_Float16)f3.w;
        // ---- B tile: element e = K = k0+kbase+e of column `col` (only cols 0,1 nonzero)
        v16h b = {};
        if (col < 2) {
            const float4* wcol = (const float4*)(gw + col * HD);
            float4 w0 = wcol[(k0 + kbase) >> 2];
            float4 w1 = wcol[((k0 + kbase) >> 2) + 1];
            float4 w2 = wcol[((k0 + kbase) >> 2) + 2];
            float4 w3 = wcol[((k0 + kbase) >> 2) + 3];
            b[0]=(_Float16)w0.x; b[1]=(_Float16)w0.y; b[2]=(_Float16)w0.z; b[3]=(_Float16)w0.w;
            b[4]=(_Float16)w1.x; b[5]=(_Float16)w1.y; b[6]=(_Float16)w1.z; b[7]=(_Float16)w1.w;
            b[8]=(_Float16)w2.x; b[9]=(_Float16)w2.y; b[10]=(_Float16)w2.z; b[11]=(_Float16)w2.w;
            b[12]=(_Float16)w3.x; b[13]=(_Float16)w3.y; b[14]=(_Float16)w3.z; b[15]=(_Float16)w3.w;
        }
        acc = __builtin_amdgcn_wmma_f32_16x16x32_f16(
            /*neg_a=*/false, a, /*neg_b=*/false, b,
            /*c_mod=*/(short)0, acc, /*reuse_a=*/false, /*reuse_b=*/false);
    }

    // D layout: lanes 0-15 hold N=lane, M=r..  lanes 16-31 hold N=lane-16, M=8+r
    int ncol = lane & 15;
    int mofs = (lane >> 4) * 8;
    if (ncol < 2) {
        float* outp = (ncol == 0) ? adst : asrc;
        int tbase = wave * 16 + mofs;
#pragma unroll
        for (int r = 0; r < 8; ++r) {
            int rr = tbase + r;
            if (rr < N) outp[rr] = acc[r];
        }
    }
}

// ---------------- edge scatter: one wave per edge, lane = float4 chunk of the row ----------------
__global__ void edge_kernel(const float* __restrict__ h,
                            const int* __restrict__ src, const int* __restrict__ dst,
                            const float* __restrict__ adst, const float* __restrict__ asrc,
                            const float* __restrict__ dn,
                            const float* __restrict__ gb, int layer,
                            float* __restrict__ mbuf, int E) {
    int wave = (blockIdx.x * blockDim.x + threadIdx.x) >> 5;
    int lane = threadIdx.x & 31;
    if (wave >= E) return;
    int s = src[wave];
    int t = dst[wave];
    s = __builtin_amdgcn_readfirstlane(s);
    t = __builtin_amdgcn_readfirstlane(t);
    float g    = tanhf(adst[t] + asrc[s] + gb[layer]);
    float coef = g * dn[t] * dn[s];
    float4 v = ((const float4*)(h + (size_t)s * HD))[lane];
    float* mp = mbuf + (size_t)t * HD + lane * 4;
    atomAddF(mp + 0, coef * v.x);
    atomAddF(mp + 1, coef * v.y);
    atomAddF(mp + 2, coef * v.z);
    atomAddF(mp + 3, coef * v.w);
}

// ---------------- h' = relu(EPS*h + m)  (out may alias m) ----------------
__global__ void update_kernel(const float* __restrict__ hin, const float* __restrict__ m,
                              float* __restrict__ out, int total4) {
    int i = blockIdx.x * blockDim.x + threadIdx.x;
    if (i >= total4) return;
    float4 a = ((const float4*)hin)[i];
    float4 b = ((const float4*)m)[i];
    float4 r;
    r.x = fmaxf(EPSW * a.x + b.x, 0.0f);
    r.y = fmaxf(EPSW * a.y + b.y, 0.0f);
    r.z = fmaxf(EPSW * a.z + b.z, 0.0f);
    r.w = fmaxf(EPSW * a.w + b.w, 0.0f);
    ((float4*)out)[i] = r;
}

extern "C" void kernel_launch(void* const* d_in, const int* in_sizes, int n_in,
                              void* d_out, int out_size, void* d_ws, size_t ws_size,
                              hipStream_t stream) {
    const float* h0  = (const float*)d_in[0];
    const int*   ei  = (const int*)d_in[1];
    const float* gw  = (const float*)d_in[2];
    const float* gb  = (const float*)d_in[3];
    const int N = in_sizes[0] / HD;
    const int E = in_sizes[1] / 2;
    const int* srcv = ei;
    const int* dstv = ei + E;

    float* ws    = (float*)d_ws;
    float* dnorm = ws;            // N
    float* adst  = ws + (size_t)N;        // N
    float* asrc  = ws + (size_t)2 * N;    // N
    float* buf   = ws + (size_t)3 * N;    // N*HD (m for layer0, then h1 in-place)
    float* out   = (float*)d_out;

    const int numTiles = (N + 15) / 16;
    dim3 blk(256);
    dim3 gridDeg((E + 255) / 256);
    dim3 gridN((N + 255) / 256);
    dim3 gridProj((numTiles * 32 + 255) / 256);
    dim3 gridEdge(((size_t)E * 32 + 255) / 256);
    int total4 = (N * HD) / 4;
    dim3 gridUpd((total4 + 255) / 256);

    // degree norm d = rsqrt(max(deg,1))
    hipMemsetAsync(dnorm, 0, (size_t)N * sizeof(float), stream);
    deg_kernel<<<gridDeg, blk, 0, stream>>>(dstv, dnorm, E);
    norm_kernel<<<gridN, blk, 0, stream>>>(dnorm, N);

    // ---- layer 0: m -> buf, then buf = relu(EPS*h0 + buf) = h1
    hipMemsetAsync(buf, 0, (size_t)N * HD * sizeof(float), stream);
    proj_kernel<<<gridProj, blk, 0, stream>>>(h0, gw, adst, asrc, N);
    edge_kernel<<<gridEdge, blk, 0, stream>>>(h0, srcv, dstv, adst, asrc, dnorm, gb, 0, buf, E);
    update_kernel<<<gridUpd, blk, 0, stream>>>(h0, buf, buf, total4);

    // ---- layer 1: m -> d_out, then d_out = relu(EPS*h1 + d_out)
    hipMemsetAsync(out, 0, (size_t)out_size * sizeof(float), stream);
    proj_kernel<<<gridProj, blk, 0, stream>>>(buf, gw + 2 * HD, adst, asrc, N);
    edge_kernel<<<gridEdge, blk, 0, stream>>>(buf, srcv, dstv, adst, asrc, dnorm, gb, 1, out, E);
    update_kernel<<<gridUpd, blk, 0, stream>>>(buf, out, out, total4);
}
